// KalmanNet_283467842221
// MI455X (gfx1250) — matrix operand order repaired and checked by reading, not compiled, not run on hardware
//
#include <hip/hip_runtime.h>
#include <hip/hip_bf16.h>

typedef __attribute__((ext_vector_type(16))) _Float16 v16h;
typedef __attribute__((ext_vector_type(8)))  _Float16 half8;
typedef __attribute__((ext_vector_type(8)))  float    v8f;

#define HIDN  64
#define INDIM 13
#define GATES 192
#define TLEN  512
#define MB    16     // batch rows per workgroup
#define NT    128    // 4 waves of 32
#define DTs   0.01f

union V16U { v16h v; half8 h[2]; };

__device__ __forceinline__ v8f wmma16(const v16h a, const v16h b, v8f c) {
    // D = A(16x32 f16) * B(32x16 f16) + C(16x16 f32)
    return __builtin_amdgcn_wmma_f32_16x16x32_f16(
        /*neg_a=*/false, a, /*neg_b=*/false, b,
        /*c_mod=*/(short)0, c, /*reuse_a=*/false, /*reuse_b=*/false);
}

__global__ __launch_bounds__(NT) void kalmannet_scan_kernel(
    const float* __restrict__ inputs,
    const float* __restrict__ ln_gamma, const float* __restrict__ ln_beta,
    const float* __restrict__ w_ih, const float* __restrict__ w_hh,
    const float* __restrict__ b_ih, const float* __restrict__ b_hh,
    const float* __restrict__ fc_w, const float* __restrict__ fc_b,
    float* __restrict__ out)
{
    // ---- LDS: weights in WMMA B-operand lane layout, resident for all 512 steps ----
    __shared__ __align__(32) _Float16 whh_B[2][12][32][16]; // 24576 B  [kblk][ntile][lane][16 f16]
    __shared__ __align__(32) _Float16 wih_B[12][16][16];    //  6144 B  lanes 0..15 only (K=0..15)
    __shared__ __align__(32) _Float16 fcw_B[2][3][32][16];  //  6144 B
    __shared__ __align__(32) _Float16 zrow[16];             //    32 B  zero K-chunk for wih lanes>=16
    __shared__ __align__(32) _Float16 hF16[MB][HIDN];       //  2048 B  GRU hidden state (f16)
    __shared__ __align__(32) _Float16 gin_s[MB][32];        //  1024 B  layernormed gru_in, K-padded
    __shared__ float g_s  [MB][128];                        //  8192 B  r|z gate preacts (gi+gh)
    __shared__ float gin_n[MB][HIDN];                       //  4096 B  i_n
    __shared__ float ghn_s[MB][HIDN];                       //  4096 B  h_n
    __shared__ float K_s  [MB][48];                         //  3072 B  Kalman gain tile
    __shared__ float bih_s[GATES], bhh_s[GATES];            //  1536 B
    __shared__ float fcb_s[48];                             //   192 B
    __shared__ float lng_s[16], lnb_s[16];                  //   128 B

    const int tid = threadIdx.x;
    const int b0  = blockIdx.x * MB;

    // ---- one-time weight staging (pre-swizzle into WMMA lane layout) ----
    for (int i = tid; i < 2*12*32*16; i += NT) {
        int e = i & 15, lane = (i >> 4) & 31, rest = i >> 9;   // rest in [0,24)
        int tile = rest % 12, kb = rest / 12;
        int n = tile*16 + (lane & 15);
        int k = kb*32 + (lane >> 4)*16 + e;
        whh_B[kb][tile][lane][e] = (_Float16)w_hh[n*HIDN + k];
    }
    for (int i = tid; i < 12*16*16; i += NT) {
        int e = i & 15, lane = (i >> 4) & 15, tile = i >> 8;
        int n = tile*16 + lane;
        wih_B[tile][lane][e] = (e < INDIM) ? (_Float16)w_ih[n*INDIM + e] : (_Float16)0.0f;
    }
    for (int i = tid; i < 2*3*32*16; i += NT) {
        int e = i & 15, lane = (i >> 4) & 31, rest = i >> 9;   // rest in [0,6)
        int tile = rest % 3, kb = rest / 3;
        int n = tile*16 + (lane & 15);
        int k = kb*32 + (lane >> 4)*16 + e;
        fcw_B[kb][tile][lane][e] = (n < 36) ? (_Float16)fc_w[n*HIDN + k] : (_Float16)0.0f;
    }
    for (int i = tid; i < GATES; i += NT) { bih_s[i] = b_ih[i]; bhh_s[i] = b_hh[i]; }
    for (int i = tid; i < 48; i += NT) fcb_s[i] = (i < 36) ? fc_b[i] : 0.0f;
    for (int i = tid; i < 16; i += NT) {
        zrow[i]  = (_Float16)0.0f;
        lng_s[i] = (i < INDIM) ? ln_gamma[i] : 0.0f;
        lnb_s[i] = (i < INDIM) ? ln_beta[i]  : 0.0f;
    }
    for (int i = tid; i < MB*HIDN; i += NT) hF16[i >> 6][i & 63] = (_Float16)0.0f;
    for (int i = tid; i < MB*32;   i += NT) gin_s[i >> 5][i & 31] = (_Float16)0.0f;
    __syncthreads();

    // ---- per-lane geometry ----
    const int wv = tid >> 5, ln = tid & 31;
    const int rowA = ln & 15, hiA = ln >> 4;   // A-operand: row, K-half
    const int colC = ln & 15, hiC = ln >> 4;   // C/D: column, row-half

    // filter state in registers of lanes 0..15 (one batch row each)
    float sp0=0,sp1=0,sp2=0, sv0=0,sv1=0,sv2=0;
    float qw=1.f,qx=0,qy=0,qz=0;
    float inn0=0,inn1=0,inn2=0,inn3=0,inn4=0,inn5=0;
    float pp0=0,pp1=0,pp2=0, vp0=0,vp1=0,vp2=0;
    float uw=1.f,ux=0,uy=0,uz=0;

    const float* inp_row = inputs + (size_t)(b0 + (tid & 15)) * TLEN * 9;
    float*       out_row = out    + (size_t)(b0 + (tid & 15)) * TLEN * 10;

    for (int t = 0; t < TLEN; ++t) {
        // ========== Phase A: quaternion physics + layernorm (lanes 0..15) ==========
        if (tid < MB) {
            const float* ip = inp_row + t*9;
            float ax=ip[0], ay=ip[1], az=ip[2];
            float gx=ip[3], gy=ip[4], gz=ip[5];
            float mx=ip[6], my=ip[7], mz=ip[8];
            if (t + 1 < TLEN) __builtin_prefetch(ip + 9, 0, 1);

            float hx=gx*(0.5f*DTs), hy=gy*(0.5f*DTs), hz=gz*(0.5f*DTs);
            float dinv = rsqrtf(1.f + hx*hx + hy*hy + hz*hz);
            float dw=dinv, dx=hx*dinv, dy=hy*dinv, dz=hz*dinv;
            uw = qw*dw - qx*dx - qy*dy - qz*dz;
            ux = qw*dx + qx*dw + qy*dz - qz*dy;
            uy = qw*dy - qx*dz + qy*dw + qz*dx;
            uz = qw*dz + qx*dy - qy*dx + qz*dw;
            float qn = rsqrtf(uw*uw + ux*ux + uy*uy + uz*uz);
            uw*=qn; ux*=qn; uy*=qn; uz*=qn;

            float xx=qx*qx, yy=qy*qy, zz=qz*qz;
            float xy=qx*qy, xz=qx*qz, yz=qy*qz;
            float wx=qw*qx, wy=qw*qy, wz=qw*qz;
            float aw0 = (1.f-2.f*(yy+zz))*ax + 2.f*(xy-wz)*ay + 2.f*(xz+wy)*az;
            float aw1 = 2.f*(xy+wz)*ax + (1.f-2.f*(xx+zz))*ay + 2.f*(yz-wx)*az;
            float aw2 = 2.f*(xz-wy)*ax + 2.f*(yz+wx)*ay + (1.f-2.f*(xx+yy))*az - 9.81f;
            vp0 = sv0 + aw0*DTs; vp1 = sv1 + aw1*DTs; vp2 = sv2 + aw2*DTs;
            pp0 = sp0 + sv0*DTs + aw0*(0.5f*DTs*DTs);
            pp1 = sp1 + sv1*DTs + aw1*(0.5f*DTs*DTs);
            pp2 = sp2 + sv2*DTs + aw2*(0.5f*DTs*DTs);

            float pxx=ux*ux, pyy=uy*uy, pzz=uz*uz;
            float pxy=ux*uy, pxz=ux*uz, pyz=uy*uz;
            float pwx=uw*ux, pwy=uw*uy, pwz=uw*uz;
            float s00=1.f-2.f*(pyy+pzz), s01=2.f*(pxy-pwz), s02=2.f*(pxz+pwy);
            float s20=2.f*(pxz-pwy),     s21=2.f*(pyz+pwx), s22=1.f-2.f*(pxx+pyy);
            // g_body = R(q_pred)^T * (0,0,9.81); mag_body = R^T * (20,0,-40)
            inn0 = mx - (20.f*s00 - 40.f*s20);
            inn1 = my - (20.f*s01 - 40.f*s21);
            inn2 = mz - (20.f*s02 - 40.f*s22);
            inn3 = ax - 9.81f*s20;
            inn4 = ay - 9.81f*s21;
            inn5 = az - 9.81f*s22;

            float gi[13] = {inn0,inn1,inn2,inn3,inn4,inn5, vp0,vp1,vp2, uw,ux,uy,uz};
            float mu = 0.f;
            #pragma unroll
            for (int i=0;i<13;++i) mu += gi[i];
            mu *= (1.f/13.f);
            float var = 0.f;
            #pragma unroll
            for (int i=0;i<13;++i) { float d = gi[i]-mu; var += d*d; }
            var *= (1.f/13.f);
            float sc = rsqrtf(var + 1e-5f);
            #pragma unroll
            for (int i=0;i<13;++i)
                gin_s[tid][i] = (_Float16)((gi[i]-mu)*sc*lng_s[i] + lnb_s[i]);
        }
        __syncthreads();

        // ========== Phase B: GRU gate GEMMs — 9 WMMAs per wave ==========
        {
            V16U agin, ah0, ah1;
            agin.h[0] = *(const half8*)&gin_s[rowA][hiA*16];
            agin.h[1] = *(const half8*)&gin_s[rowA][hiA*16 + 8];
            ah0.h[0]  = *(const half8*)&hF16[rowA][hiA*16];
            ah0.h[1]  = *(const half8*)&hF16[rowA][hiA*16 + 8];
            ah1.h[0]  = *(const half8*)&hF16[rowA][32 + hiA*16];
            ah1.h[1]  = *(const half8*)&hF16[rowA][32 + hiA*16 + 8];

            const int tr = wv, tz = wv + 4, tn = wv + 8;
            v16h bir = (ln < 16) ? *(const v16h*)&wih_B[tr][ln][0] : *(const v16h*)&zrow[0];
            v16h biz = (ln < 16) ? *(const v16h*)&wih_B[tz][ln][0] : *(const v16h*)&zrow[0];
            v16h bin = (ln < 16) ? *(const v16h*)&wih_B[tn][ln][0] : *(const v16h*)&zrow[0];

            v8f accR = {};
            accR = wmma16(agin.v, bir, accR);
            accR = wmma16(ah0.v, *(const v16h*)&whh_B[0][tr][ln][0], accR);
            accR = wmma16(ah1.v, *(const v16h*)&whh_B[1][tr][ln][0], accR);
            v8f accZ = {};
            accZ = wmma16(agin.v, biz, accZ);
            accZ = wmma16(ah0.v, *(const v16h*)&whh_B[0][tz][ln][0], accZ);
            accZ = wmma16(ah1.v, *(const v16h*)&whh_B[1][tz][ln][0], accZ);
            v8f accI = {};
            accI = wmma16(agin.v, bin, accI);
            v8f accH = {};
            accH = wmma16(ah0.v, *(const v16h*)&whh_B[0][tn][ln][0], accH);
            accH = wmma16(ah1.v, *(const v16h*)&whh_B[1][tn][ln][0], accH);

            #pragma unroll
            for (int r = 0; r < 8; ++r) {
                int mrow = r + 8*hiC;
                g_s  [mrow][wv*16 + colC]      = accR[r];
                g_s  [mrow][64 + wv*16 + colC] = accZ[r];
                gin_n[mrow][wv*16 + colC]      = accI[r];
                ghn_s[mrow][wv*16 + colC]      = accH[r];
            }
        }
        __syncthreads();

        // ========== Phase C: GRU elementwise update (all 128 threads, 8 elems each) ==========
        #pragma unroll
        for (int e = 0; e < 8; ++e) {
            int idx = tid*8 + e;               // 0..1023 over 16x64
            int mrow = idx >> 6, j = idx & 63;
            float gr  = g_s[mrow][j]        + bih_s[j]        + bhh_s[j];
            float gz2 = g_s[mrow][64 + j]   + bih_s[64 + j]   + bhh_s[64 + j];
            float gin = gin_n[mrow][j]      + bih_s[128 + j];
            float ghn = ghn_s[mrow][j]      + bhh_s[128 + j];
            float rg = 1.f / (1.f + __expf(-gr));
            float zg = 1.f / (1.f + __expf(-gz2));
            float ng = tanhf(gin + rg * ghn);
            float hp = (float)hF16[mrow][j];
            hF16[mrow][j] = (_Float16)((1.f - zg) * ng + zg * hp);
        }
        __syncthreads();

        // ========== Phase D: Kalman-gain FC GEMM (waves 0..2, 2 WMMAs each) ==========
        if (wv < 3) {
            V16U ah0, ah1;
            ah0.h[0] = *(const half8*)&hF16[rowA][hiA*16];
            ah0.h[1] = *(const half8*)&hF16[rowA][hiA*16 + 8];
            ah1.h[0] = *(const half8*)&hF16[rowA][32 + hiA*16];
            ah1.h[1] = *(const half8*)&hF16[rowA][32 + hiA*16 + 8];
            v8f accK = {};
            accK = wmma16(ah0.v, *(const v16h*)&fcw_B[0][wv][ln][0], accK);
            accK = wmma16(ah1.v, *(const v16h*)&fcw_B[1][wv][ln][0], accK);
            #pragma unroll
            for (int r = 0; r < 8; ++r)
                K_s[r + 8*hiC][wv*16 + colC] = accK[r];
        }
        __syncthreads();

        // ========== Phase E: correction + output + state update (lanes 0..15) ==========
        if (tid < MB) {
            float inn[6] = {inn0,inn1,inn2,inn3,inn4,inn5};
            float corr[6];
            #pragma unroll
            for (int i = 0; i < 6; ++i) {
                float c = 0.f;
                #pragma unroll
                for (int j = 0; j < 6; ++j)
                    c += (K_s[tid][i*6 + j] + fcb_s[i*6 + j]) * inn[j];
                corr[i] = c;
            }
            sp0 = pp0 + corr[0]; sp1 = pp1 + corr[1]; sp2 = pp2 + corr[2];
            sv0 = vp0 + corr[3]; sv1 = vp1 + corr[4]; sv2 = vp2 + corr[5];
            qw = uw; qx = ux; qy = uy; qz = uz;
            float* op = out_row + t*10;
            op[0]=sp0; op[1]=sp1; op[2]=sp2;
            op[3]=sv0; op[4]=sv1; op[5]=sv2;
            op[6]=qw;  op[7]=qx;  op[8]=qy;  op[9]=qz;
        }
        // no barrier needed: next writer of K_s (phase D) is >=3 barriers away
    }
}

extern "C" void kernel_launch(void* const* d_in, const int* in_sizes, int n_in,
                              void* d_out, int out_size, void* d_ws, size_t ws_size,
                              hipStream_t stream) {
    const float* inputs   = (const float*)d_in[0];
    const float* ln_gamma = (const float*)d_in[1];
    const float* ln_beta  = (const float*)d_in[2];
    const float* w_ih     = (const float*)d_in[3];
    const float* w_hh     = (const float*)d_in[4];
    const float* b_ih     = (const float*)d_in[5];
    const float* b_hh     = (const float*)d_in[6];
    const float* fc_w     = (const float*)d_in[7];
    const float* fc_b     = (const float*)d_in[8];
    float* out            = (float*)d_out;
    (void)d_ws; (void)ws_size; (void)n_in; (void)out_size;

    int Btot = in_sizes[0] / (TLEN * 9);   // 4096
    int nblocks = Btot / MB;               // 256 workgroups x 128 threads (4 waves)
    kalmannet_scan_kernel<<<nblocks, NT, 0, stream>>>(
        inputs, ln_gamma, ln_beta, w_ih, w_hh, b_ih, b_hh, fc_w, fc_b, out);
}
